// KANPolyLayer_70463233458501
// MI455X (gfx1250) — compile-verified, matrix operand order checked by reading, not run on hardware
//
#include <hip/hip_runtime.h>

// ---- CDNA5 vector types -----------------------------------------------------
typedef __attribute__((ext_vector_type(16))) _Float16 v16h;
typedef __attribute__((ext_vector_type(8)))  _Float16 v8h;   // 16 bytes
typedef __attribute__((ext_vector_type(4)))  _Float16 v4h;   // 8 bytes
typedef __attribute__((ext_vector_type(8)))  float    v8f;

union Frag { v16h v; v8h h[2]; };

// Problem dims (fixed by the reference)
#define M_DIM 4096
#define N_DIM 1024
#define K_DIM 1024   // input features per power plane
#define NPOW  4      // powers 1..4 (power 0 folded into bias)

#define BM 128
#define BN 128
#define BK 32        // WMMA K for f16
#define NIT (K_DIM / BK)

// ---------------------------------------------------------------------------
// Kernel 1: repack coeffs[o][i][p] (f32, p-interleaved) into 4 contiguous f16
// planes Wb[p][o][i]; fold p=0 plane + bias into bias_eff[o].
// ---------------------------------------------------------------------------
__global__ void kan_prepack(const float* __restrict__ coeffs,
                            const float* __restrict__ bias,
                            _Float16* __restrict__ Wb,
                            float* __restrict__ bias_eff) {
    const int o = blockIdx.x;      // 0..1023 (output channel)
    const int t = threadIdx.x;     // 0..255
    float c0sum = 0.0f;
    for (int i = t; i < K_DIM; i += 256) {
        const float* c = coeffs + (size_t)(o * K_DIM + i) * 5;
        c0sum += c[0];
        Wb[(size_t)(0 * N_DIM + o) * K_DIM + i] = (_Float16)c[1];
        Wb[(size_t)(1 * N_DIM + o) * K_DIM + i] = (_Float16)c[2];
        Wb[(size_t)(2 * N_DIM + o) * K_DIM + i] = (_Float16)c[3];
        Wb[(size_t)(3 * N_DIM + o) * K_DIM + i] = (_Float16)c[4];
    }
    __shared__ float red[256];
    red[t] = c0sum;
    __syncthreads();
    #pragma unroll
    for (int s = 128; s > 0; s >>= 1) {
        if (t < s) red[t] += red[t + s];
        __syncthreads();
    }
    if (t == 0) bias_eff[o] = bias[o] + red[0];
}

// ---------------------------------------------------------------------------
// Kernel 2: y = sum_{p=1..4} (x^p) @ Wb[p]^T + bias_eff   via v_wmma f16->f32
//   Block: 256 threads = 8 waves. Block tile 128x128, double-buffered LDS.
//   Wave tile 32(M) x 64(N) = 2x4 WMMA 16x16 accumulators.
//   B tiles staged with GLOBAL_LOAD_ASYNC_TO_LDS_B128 (ASYNCcnt pipeline).
// ---------------------------------------------------------------------------
__global__ void __launch_bounds__(256)
kan_wmma_gemm(const float* __restrict__ x,
              const _Float16* __restrict__ Wb,
              const float* __restrict__ bias_eff,
              float* __restrict__ out) {
    // Double buffers: 2 x (4 power planes A[128x32] + 4 weight planes B[128x32])
    __shared__ __align__(16) _Float16 As[2][NPOW * BM * BK]; // 2 x 32 KB
    __shared__ __align__(16) _Float16 Bs[2][NPOW * BN * BK]; // 2 x 32 KB

    const int t    = threadIdx.x;
    const int lane = t & 31;
    const int wave = t >> 5;                 // 0..7
    const int wm   = (wave >> 1) * 32;       // wave M offset in block tile
    const int wn   = (wave & 1) * 64;        // wave N offset in block tile
    const int m0   = blockIdx.x * BM;
    const int n0   = blockIdx.y * BN;

    const int half = lane >> 4;              // K-chunk select per WMMA layout
    const int fr   = lane & 15;              // fragment row (M or N within tile)

    v8f acc[2][4];
    #pragma unroll
    for (int mt = 0; mt < 2; ++mt)
        #pragma unroll
        for (int nt = 0; nt < 4; ++nt)
            acc[mt][nt] = (v8f){0.f, 0.f, 0.f, 0.f, 0.f, 0.f, 0.f, 0.f};

    const unsigned long long wbBase = (unsigned long long)(__UINTPTR_TYPE__)Wb;

    // ---- B staging: pure copy -> async global->LDS (GVS mode, b128) --------
    auto stageB = [&](int k0, int buf) {
        const unsigned bsBase = (unsigned)(__UINTPTR_TYPE__)(&Bs[buf][0]);
        #pragma unroll
        for (int j = 0; j < 8; ++j) {           // 2048 16B units / 256 threads
            const int u     = j * 256 + t;
            const int plane = u >> 9;           // 512 units per plane
            const int rem   = u & 511;
            const int n     = rem >> 2;         // 4 units per 32-f16 row
            const int kq    = rem & 3;
            const unsigned goff =
                (unsigned)(((plane * N_DIM + n0 + n) * K_DIM + k0 + kq * 8) *
                           (int)sizeof(_Float16));
            const unsigned loff =
                bsBase + (unsigned)(((plane * BN + n) * BK + kq * 8) *
                                    (int)sizeof(_Float16));
            asm volatile("global_load_async_to_lds_b128 %0, %1, %2"
                         :: "v"(loff), "v"(goff), "s"(wbBase)
                         : "memory");
        }
    };

    // ---- A staging: load x (f32), expand powers 1..4 to f16 in LDS ---------
    auto stageA = [&](int k0, int buf) {
        _Float16* a = &As[buf][0];
        #pragma unroll
        for (int r = 0; r < 4; ++r) {           // 1024 float4 units / 256 thr
            const int row = r * 32 + (t >> 3);
            const int col = (t & 7) * 4;
            const float4 xv =
                *(const float4*)(x + (size_t)(m0 + row) * K_DIM + k0 + col);
            v4h p1, p2, p3, p4;
            const float e[4] = {xv.x, xv.y, xv.z, xv.w};
            #pragma unroll
            for (int j = 0; j < 4; ++j) {
                const float x1 = e[j];
                const float x2 = x1 * x1;
                const float x3 = x2 * x1;
                const float x4 = x2 * x2;
                p1[j] = (_Float16)x1;
                p2[j] = (_Float16)x2;
                p3[j] = (_Float16)x3;
                p4[j] = (_Float16)x4;
            }
            *(v4h*)&a[(0 * BM + row) * BK + col] = p1;
            *(v4h*)&a[(1 * BM + row) * BK + col] = p2;
            *(v4h*)&a[(2 * BM + row) * BK + col] = p3;
            *(v4h*)&a[(3 * BM + row) * BK + col] = p4;
        }
    };

    // ---- Prologue: stage tile 0 --------------------------------------------
    stageB(0, 0);
    stageA(0, 0);
    asm volatile("s_wait_asynccnt 0x0" ::: "memory");
    __syncthreads();

    // ---- Pipelined main loop: stage k+1 while computing k ------------------
    for (int it = 0; it < NIT; ++it) {
        const int buf = it & 1;
        if (it + 1 < NIT) {
            stageB((it + 1) * BK, buf ^ 1);
            stageA((it + 1) * BK, buf ^ 1);
        }

        const _Float16* ab = &As[buf][0];
        const _Float16* bb = &Bs[buf][0];
        #pragma unroll
        for (int p = 0; p < NPOW; ++p) {
            Frag a[2], b[4];
            #pragma unroll
            for (int mt = 0; mt < 2; ++mt) {
                const _Float16* base = &ab[((p * BM) + wm + mt * 16 + fr) * BK];
                a[mt].h[0] = *(const v8h*)(base + half * 8);      // K 0-7 / 8-15
                a[mt].h[1] = *(const v8h*)(base + 16 + half * 8); // K 16-23 / 24-31
            }
            #pragma unroll
            for (int nt = 0; nt < 4; ++nt) {
                const _Float16* base = &bb[((p * BN) + wn + nt * 16 + fr) * BK];
                b[nt].h[0] = *(const v8h*)(base + half * 8);
                b[nt].h[1] = *(const v8h*)(base + 16 + half * 8);
            }
            #pragma unroll
            for (int mt = 0; mt < 2; ++mt)
                #pragma unroll
                for (int nt = 0; nt < 4; ++nt)
                    acc[mt][nt] = __builtin_amdgcn_wmma_f32_16x16x32_f16(
                        false, a[mt].v, false, b[nt].v,
                        (short)0, acc[mt][nt], false, false);
        }

        // Tile k+1 (async B loads + DS A stores) must be complete and visible
        // to all waves before the next compute phase.
        asm volatile("s_wait_asynccnt 0x0" ::: "memory");
        __syncthreads();
    }

    // ---- Epilogue: add bias_eff, store f32 per C-matrix lane layout --------
    #pragma unroll
    for (int mt = 0; mt < 2; ++mt) {
        #pragma unroll
        for (int nt = 0; nt < 4; ++nt) {
            const int col  = n0 + wn + nt * 16 + fr;
            const int mrow = m0 + wm + mt * 16 + half * 8;
            const float bv = bias_eff[col];
            #pragma unroll
            for (int r = 0; r < 8; ++r)
                out[(size_t)(mrow + r) * N_DIM + col] = acc[mt][nt][r] + bv;
        }
    }
}

// ---------------------------------------------------------------------------
extern "C" void kernel_launch(void* const* d_in, const int* in_sizes, int n_in,
                              void* d_out, int out_size, void* d_ws, size_t ws_size,
                              hipStream_t stream) {
    const float* x      = (const float*)d_in[0];   // (4096, 1024) f32
    const float* coeffs = (const float*)d_in[1];   // (1024, 1024, 5) f32
    const float* bias   = (const float*)d_in[2];   // (1, 1024) f32
    float* out          = (float*)d_out;           // (4096, 1024) f32

    // Workspace: 4 f16 weight planes (8 MB) + effective bias (4 KB)
    _Float16* Wb    = (_Float16*)d_ws;
    float* bias_eff = (float*)((char*)d_ws + (size_t)NPOW * N_DIM * K_DIM * sizeof(_Float16));

    kan_prepack<<<dim3(N_DIM), dim3(256), 0, stream>>>(coeffs, bias, Wb, bias_eff);

    dim3 grid(M_DIM / BM, N_DIM / BN);  // 32 x 8
    kan_wmma_gemm<<<grid, dim3(256), 0, stream>>>(x, Wb, bias_eff, out);
}
